// ScaleAwareDynamicConv2d_15685220565220
// MI455X (gfx1250) — matrix operand order repaired and checked by reading, not compiled, not run on hardware
//
#include <hip/hip_runtime.h>
#include <hip/hip_bf16.h>
#include <stdint.h>

typedef __attribute__((ext_vector_type(16))) __bf16   v16bf;
typedef __attribute__((ext_vector_type(8)))  float    v8f;
typedef __attribute__((ext_vector_type(8)))  uint32_t v8u;

#define BS   32
#define CC   64
#define HH   128
#define WW   128
#define OO   64
#define NKER 4
#define HID  17
#define PH   130   // padded H (1-pixel halo)
#define PW   130   // padded W

// workspace layout (bytes)
static constexpr size_t OFF_POOLED = 0;                    // 8 KB
static constexpr size_t OFF_ATTN   = 8192;                 // 512 B
static constexpr size_t OFF_AGGB   = 16384;                // 8 KB
static constexpr size_t OFF_AGGW   = 65536;                // 2.25 MB (bf16, fragment-swizzled)
static constexpr size_t OFF_XT     = (size_t)4 << 20;      // 32*130*130*64*2 = 66 MB (padded NHWC bf16)

__device__ __forceinline__ uint16_t f32_to_bf16(float f) {
  uint32_t u = __float_as_uint(f);
  u += 0x7FFFu + ((u >> 16) & 1u);     // round to nearest even
  return (uint16_t)(u >> 16);
}

// ---------------------------------------------------------------------------
// 1) pooled[b,c] = mean_{h,w} x[b,c,h,w]
// ---------------------------------------------------------------------------
__global__ __launch_bounds__(256) void pool_kernel(const float* __restrict__ x,
                                                   float* __restrict__ pooled) {
  const int bc = blockIdx.x;
  const float* p = x + (size_t)bc * (HH * WW);
  float s = 0.f;
  for (int i = threadIdx.x; i < HH * WW; i += 256) s += p[i];
  __shared__ float red[256];
  red[threadIdx.x] = s;
  __syncthreads();
  for (int st = 128; st > 0; st >>= 1) {
    if (threadIdx.x < (unsigned)st) red[threadIdx.x] += red[threadIdx.x + st];
    __syncthreads();
  }
  if (threadIdx.x == 0) pooled[bc] = red[0] * (1.0f / (HH * WW));
}

// ---------------------------------------------------------------------------
// 2a) zero the 1-pixel halo ring of padded NHWC buffer. grid: (BS), block 256
// ---------------------------------------------------------------------------
__global__ __launch_bounds__(256) void halo_kernel(uint16_t* __restrict__ xt) {
  const int b = blockIdx.x;
  uint32_t* p = (uint32_t*)(xt + (size_t)b * PH * PW * CC);   // row stride = PW*CC/2 = 4160 u32
  const int ROWU = PW * CC / 2;
  for (int i = threadIdx.x; i < ROWU; i += 256) {
    p[i] = 0u;                                 // y = 0
    p[(size_t)(PH - 1) * ROWU + i] = 0u;       // y = 129
  }
  for (int i = threadIdx.x; i < HH * (CC / 2); i += 256) {
    const int y = 1 + (i >> 5), j = i & 31;    // 32 u32 per pixel
    p[(size_t)y * ROWU + j] = 0u;              // x = 0
    p[(size_t)y * ROWU + (PW - 1) * (CC / 2) + j] = 0u;  // x = 129
  }
}

// ---------------------------------------------------------------------------
// 2b) NCHW f32 -> padded NHWC bf16 (LDS tile transpose). grid: (W/64, H, B)
// ---------------------------------------------------------------------------
__global__ __launch_bounds__(256) void nhwc_kernel(const float* __restrict__ x,
                                                   uint16_t* __restrict__ xt) {
  const int b = blockIdx.z, y = blockIdx.y, x0 = blockIdx.x * 64;
  __shared__ uint16_t lds[64][72];
  const int t  = threadIdx.x;
  const int xx = t & 63, cb = t >> 6;
#pragma unroll
  for (int co = 0; co < 16; ++co) {
    const int c = co * 4 + cb;
    const float v = x[(((size_t)b * CC + c) * HH + y) * WW + x0 + xx];
    lds[xx][c] = f32_to_bf16(v);
  }
  __syncthreads();
  const int pix = t >> 2, c0 = (t & 3) * 16;
  v8u ov;
#pragma unroll
  for (int j = 0; j < 8; ++j)
    ov[j] = (uint32_t)lds[pix][c0 + 2 * j] | ((uint32_t)lds[pix][c0 + 2 * j + 1] << 16);
  // padded destination: (y+1, x+1)
  v8u* dst = (v8u*)(xt + ((((size_t)b * PH + y + 1) * PW + x0 + pix + 1) * CC + c0));
  *dst = ov;
}

// ---------------------------------------------------------------------------
// 3) attention + aggregated bias (32 threads, one per sample)
// ---------------------------------------------------------------------------
__global__ void attn_kernel(const float* __restrict__ pooled,
                            const float* __restrict__ scale,
                            const float* __restrict__ fc1_w,
                            const float* __restrict__ fc2_w,
                            const float* __restrict__ fc2_b,
                            const float* __restrict__ bias,
                            float* __restrict__ attn,
                            float* __restrict__ aggb) {
  const int b = threadIdx.x;
  if (b >= BS) return;
  float h[HID + 2];
  for (int i = 0; i < HID; ++i) {
    float s = 0.f;
    for (int c = 0; c < CC; ++c) s += pooled[b * CC + c] * fc1_w[i * CC + c];
    h[i] = fmaxf(s, 0.f);
  }
  h[HID] = scale[0];
  h[HID + 1] = scale[0];
  float lg[NKER], mx = -3.0e38f;
  for (int k = 0; k < NKER; ++k) {
    float s = fc2_b[k];
    for (int i = 0; i < HID + 2; ++i) s += h[i] * fc2_w[k * (HID + 2) + i];
    lg[k] = s;
    mx = fmaxf(mx, s);
  }
  float den = 0.f;
  for (int k = 0; k < NKER; ++k) { lg[k] = __expf(lg[k] - mx); den += lg[k]; }
  const float inv = 1.0f / den;
  for (int k = 0; k < NKER; ++k) { lg[k] *= inv; attn[b * NKER + k] = lg[k]; }
  for (int o = 0; o < OO; ++o) {
    float s = 0.f;
    for (int k = 0; k < NKER; ++k) s += lg[k] * bias[k * OO + o];
    aggb[b * OO + o] = s;
  }
}

// ---------------------------------------------------------------------------
// 4) agg_w -> WMMA A-fragment order (per b: 4 o-tiles x 9 taps x 2 c-chunks,
//    32 lanes x 32 B contiguous per lane). grid: B*9 x 256
// ---------------------------------------------------------------------------
__global__ __launch_bounds__(256) void aggw_kernel(const float* __restrict__ weight,
                                                   const float* __restrict__ attn,
                                                   uint32_t* __restrict__ aggw) {
  const int b   = blockIdx.x / 9;
  const int fid = (blockIdx.x % 9) * 256 + threadIdx.x;
  const int f    = fid >> 5;
  const int lane = fid & 31;
  const int m = f / 18, rem = f % 18, p = rem >> 1, q = rem & 1;
  const int kh = p / 3, kw = p % 3;
  const int o  = m * 16 + (lane & 15);
  const int hi = (lane >> 4) & 1;
  const float a0 = attn[b * NKER + 0], a1 = attn[b * NKER + 1];
  const float a2 = attn[b * NKER + 2], a3 = attn[b * NKER + 3];
  v8u ov;
#pragma unroll
  for (int e2 = 0; e2 < 8; ++e2) {
    uint32_t packed = 0;
#pragma unroll
    for (int sub = 0; sub < 2; ++sub) {
      const int e = e2 * 2 + sub;
      const int K = hi * 16 + e;   // A 16x32 bf16 lane layout (contiguous k = 16*hi + e)
      const int c = q * 32 + K;
      const size_t base = (size_t)o * (CC * 9) + (size_t)c * 9 + kh * 3 + kw;
      const size_t ks = (size_t)OO * CC * 9;
      const float v = a0 * weight[base] + a1 * weight[base + ks] +
                      a2 * weight[base + 2 * ks] + a3 * weight[base + 3 * ks];
      packed |= (uint32_t)f32_to_bf16(v) << (16 * sub);
    }
    ov[e2] = packed;
  }
  v8u* dst = (v8u*)(aggw + ((size_t)b * 73728 + ((size_t)f * 32 + lane) * 32) / 4);
  *dst = ov;
}

// ---------------------------------------------------------------------------
// 5) conv: 9 shifted GEMMs, bf16 WMMA, f32 accum. Halo'd input => no bounds
//    checks, one per-lane base pointer + immediate offsets for all 36 B loads.
//    grid: (W/64, H, B), block 256 (8 waves).
//    wave: o-tile m = w&3 (16 chans) x 32 pixels (two 16-wide subtiles).
// ---------------------------------------------------------------------------
__global__ __launch_bounds__(256) void conv_kernel(const uint16_t* __restrict__ xt,
                                                   const v8u* __restrict__ aggw,
                                                   const float* __restrict__ aggb,
                                                   float* __restrict__ out) {
  const int b    = blockIdx.z;
  const int y    = blockIdx.y;
  const int wave = threadIdx.x >> 5;
  const int lane = threadIdx.x & 31;
  const int m    = wave & 3;
  const int pt   = wave >> 2;                 // 0..1
  const int x0   = blockIdx.x * 64 + pt * 32; // this wave: pixels x0 .. x0+31
  const int hi   = lane >> 4;
  const int ln   = lane & 15;

  // hoist all 18 A fragments (9 taps x 2 c-chunks) into registers
  const v8u* abase = aggw + (size_t)b * 2304 + (size_t)m * 18 * 32;
  v16bf A[18];
#pragma unroll
  for (int f = 0; f < 18; ++f)
    A[f] = __builtin_bit_cast(v16bf, abase[f * 32 + lane]);

  // per-lane B base: padded coords, corner tap (dy=-1,dx=-1) of pixel x0+ln
  const uint16_t* bptr = xt + (((size_t)b * PH + y) * PW + x0 + ln) * CC + hi * 16;

  v8f acc0 = (v8f)0.0f;
  v8f acc1 = (v8f)0.0f;

#pragma unroll
  for (int p = 0; p < 9; ++p) {
    const int dy = p / 3, dx = p % 3;                 // 0..2 in padded coords
    const int off = (dy * PW + dx) * CC;              // halfs; compile-time const
#pragma unroll
    for (int q = 0; q < 2; ++q) {
      v16bf B0 = __builtin_bit_cast(v16bf, *(const v8u*)(bptr + off + q * 32));
      v16bf B1 = __builtin_bit_cast(v16bf, *(const v8u*)(bptr + off + 16 * CC + q * 32));
      acc0 = __builtin_amdgcn_wmma_f32_16x16x32_bf16(false, A[p * 2 + q], false, B0,
                                                     (short)0, acc0, false, false);
      acc1 = __builtin_amdgcn_wmma_f32_16x16x32_bf16(false, A[p * 2 + q], false, B1,
                                                     (short)0, acc1, false, false);
    }
  }

  // C/D layout: VGPR r, lanes0-15 -> M=r, lanes16-31 -> M=8+r; N = lane&15
#pragma unroll
  for (int r = 0; r < 8; ++r) {
    const int o = m * 16 + hi * 8 + r;
    const float bb = aggb[b * OO + o];
    const size_t rowbase = (((size_t)b * OO + o) * HH + y) * WW;
    out[rowbase + x0 + ln]      = acc0[r] + bb;
    out[rowbase + x0 + 16 + ln] = acc1[r] + bb;
  }
}

// ---------------------------------------------------------------------------
extern "C" void kernel_launch(void* const* d_in, const int* in_sizes, int n_in,
                              void* d_out, int out_size, void* d_ws, size_t ws_size,
                              hipStream_t stream) {
  const float* x      = (const float*)d_in[0];
  const float* scale  = (const float*)d_in[1];
  const float* fc1_w  = (const float*)d_in[2];
  const float* fc2_w  = (const float*)d_in[3];
  const float* fc2_b  = (const float*)d_in[4];
  const float* weight = (const float*)d_in[5];
  const float* bias   = (const float*)d_in[6];
  float* out = (float*)d_out;

  char* ws = (char*)d_ws;
  float*    pooled = (float*)(ws + OFF_POOLED);
  float*    attn   = (float*)(ws + OFF_ATTN);
  float*    aggb   = (float*)(ws + OFF_AGGB);
  uint32_t* aggw   = (uint32_t*)(ws + OFF_AGGW);
  uint16_t* xt     = (uint16_t*)(ws + OFF_XT);

  pool_kernel<<<BS * CC, 256, 0, stream>>>(x, pooled);
  halo_kernel<<<BS, 256, 0, stream>>>(xt);
  nhwc_kernel<<<dim3(WW / 64, HH, BS), 256, 0, stream>>>(x, xt);
  attn_kernel<<<1, 32, 0, stream>>>(pooled, scale, fc1_w, fc2_w, fc2_b, bias, attn, aggb);
  aggw_kernel<<<BS * 9, 256, 0, stream>>>(weight, attn, aggw);
  conv_kernel<<<dim3(WW / 64, HH, BS), 256, 0, stream>>>(xt, (const v8u*)aggw, aggb, out);
}
